// Concept_network_8924942041747
// MI455X (gfx1250) — compile-verified, matrix operand order checked by reading, not run on hardware
//
#include <hip/hip_runtime.h>

// ---------------- problem constants ----------------
static constexpr int C_NOBJ = 4096;
static constexpr int C_NV   = 2048;
static constexpr int C_K    = 8;
static constexpr int C_L    = 10;
static constexpr int C_D    = 300;
static constexpr int C_F    = 512;
static constexpr int C_TM   = 3;
static constexpr int C_B    = C_NV * C_K;   // 16384 sequences
static constexpr int C_3D   = 3 * C_D;      // 900

// ---------------- types ----------------
typedef __attribute__((ext_vector_type(16))) __bf16        v16bf;
typedef __attribute__((ext_vector_type(8)))  float         v8f;
typedef __attribute__((ext_vector_type(8)))  unsigned int  v8u;
typedef __attribute__((ext_vector_type(4)))  unsigned int  v4u;
typedef __attribute__((ext_vector_type(8)))  int           v8i;
typedef __attribute__((ext_vector_type(4)))  int           v4i;
typedef unsigned short ushort_t;

__device__ __forceinline__ ushort_t cn_f32_bf16(float f) {
  unsigned int u = __builtin_bit_cast(unsigned int, f);
  unsigned int r = u + 0x7FFFu + ((u >> 16) & 1u);   // round-to-nearest-even
  return (ushort_t)(r >> 16);
}
__device__ __forceinline__ float cn_sigmoid(float x) { return 1.f / (1.f + __expf(-x)); }

// ----------------------------------------------------------------------
// TDM: async 2D tile load (64 rows x 32 bf16) global -> LDS; +1 dword pad
// per 16-dword (64B) row => LDS row stride 17 dwords.  D# per ISA §8.
// ----------------------------------------------------------------------
__device__ __forceinline__ void cn_tdm_load_tile(const ushort_t* src, long long rowStrideElems,
                                                 int rowsAvail, int colsAvail,
                                                 unsigned ldsByteOff) {
  unsigned long long ga = (unsigned long long)(uintptr_t)src;
  unsigned td0 = (unsigned)(colsAvail > 0 ? colsAvail : 0);   // tensor_dim0 (elements)
  unsigned td1 = (unsigned)(rowsAvail > 0 ? rowsAvail : 0);   // tensor_dim1 (rows)
  unsigned long long st0 = (unsigned long long)rowStrideElems;

  v4u g0;
  g0[0] = 1u;                                               // count=1, user mode
  g0[1] = ldsByteOff;                                       // lds_addr
  g0[2] = (unsigned)(ga & 0xFFFFFFFFu);                     // global_addr[31:0]
  g0[3] = (unsigned)((ga >> 32) & 0x1FFFFFFu) | (2u << 30); // addr[56:32] | type=2

  v8i g1;
  g1[0] = (int)((1u << 16) | (1u << 20) | (3u << 22));      // data_size=2B, pad_en, pad per 16 dw, pad 1 dw
  g1[1] = (int)(td0 << 16);                                 // barrier_addr=0 | tensor_dim0[15:0]
  g1[2] = (int)((td0 >> 16) | (td1 << 16));                 // tensor_dim0[31:16] | tensor_dim1[15:0]
  g1[3] = (int)((td1 >> 16) | (32u << 16));                 // tensor_dim1[31:16] | tile_dim0=32
  g1[4] = 64;                                               // tile_dim1=64 | tile_dim2=0
  g1[5] = (int)(unsigned)(st0 & 0xFFFFFFFFu);               // tensor_dim0_stride[31:0]
  g1[6] = (int)(unsigned)((st0 >> 32) & 0xFFFFu);           // stride[47:32] | dim1_stride[15:0]=0
  g1[7] = 0;

  v4i z4 = {0, 0, 0, 0};
#if __clang_major__ >= 23
  v8i z8 = {0, 0, 0, 0, 0, 0, 0, 0};
  __builtin_amdgcn_tensor_load_to_lds(g0, g1, z4, z4, z8, 0);
#else
  __builtin_amdgcn_tensor_load_to_lds(g0, g1, z4, z4, 0);
#endif
}

// ======================================================================
// GEMM:  C(M,N) = act( A(M,Kd bf16; row stride lda) * W(N,Kd bf16)^T + bias )
// f32 accumulation.  ACT: 0=none 1=tanh 2=relu
// 128 threads (4 waves), 64x64 tile, K-chunk 32, TDM double-buffered LDS.
// ======================================================================
static constexpr unsigned LDS_TILE_B  = 64 * 17 * 4;        // 4352 B per tile
static constexpr unsigned LDS_BUF_B   = 2 * LDS_TILE_B;     // 8704 B per buffer pair
static constexpr unsigned LDS_TOTAL_B = 2 * LDS_BUF_B;      // 17408 B

extern __shared__ unsigned char cn_smem[];

template <int ACT>
__global__ __launch_bounds__(128)
void cn_gemm_wmma(const ushort_t* __restrict__ A, int lda,
                  const ushort_t* __restrict__ W,
                  const float* __restrict__ bias,
                  float* __restrict__ C, int ldc,
                  int M, int N, int Kd)
{
  const int tid  = threadIdx.x;
  const int lane = tid & 31;
  const int wave = tid >> 5;             // 0..3 -> M sub-tile
  const int bm   = blockIdx.y * 64;
  const int bn   = blockIdx.x * 64;
  const int mloc = lane & 15;
  const int khal = lane >> 4;

  const unsigned int* lds = (const unsigned int*)cn_smem;
  v8f acc[4] = {};

  // prologue: wave 0 DMAs chunk 0 into buffer 0 (A then B; TENSORcnt in-order)
  if (wave == 0) {
    cn_tdm_load_tile(A + (long long)bm * lda, lda, M - bm, Kd, 0u);
    cn_tdm_load_tile(W + (long long)bn * Kd, Kd, N - bn, Kd, LDS_TILE_B);
  }

  const int nchunks = (Kd + 31) >> 5;
  for (int c = 0; c < nchunks; ++c) {
    const int ib = c & 1;
    if (wave == 0) {
      const int kn = (c + 1) << 5;
      if (kn < Kd) {
        const unsigned nb = (unsigned)(ib ^ 1) * LDS_BUF_B;
        cn_tdm_load_tile(A + (long long)bm * lda + kn, lda, M - bm, Kd - kn, nb);
        cn_tdm_load_tile(W + (long long)bn * Kd + kn, Kd, N - bn, Kd - kn, nb + LDS_TILE_B);
        __builtin_amdgcn_s_wait_tensorcnt(2);   // current chunk's pair complete
      } else {
        __builtin_amdgcn_s_wait_tensorcnt(0);
      }
    }
    __syncthreads();   // publish buf[ib] to all waves

    const unsigned int* uA = lds + (size_t)ib * (LDS_BUF_B / 4);
    const unsigned int* uB = uA + (LDS_TILE_B / 4);

    // A fragment (row = wave*16 + mloc): raw bf16 pairs, no conversion.
    // dword index within row = (v>>2)*8 + khal*4 + (v&3)
    v8u au;
#pragma unroll
    for (int v = 0; v < 8; ++v) {
      const int cd = ((v >> 2) << 3) + (khal << 2) + (v & 3);
      au[v] = uA[(wave * 16 + mloc) * 17 + cd];
    }
    const v16bf aF = __builtin_bit_cast(v16bf, au);

#pragma unroll
    for (int nt = 0; nt < 4; ++nt) {
      v8u bu;
#pragma unroll
      for (int v = 0; v < 8; ++v) {
        const int cd = ((v >> 2) << 3) + (khal << 2) + (v & 3);
        bu[v] = uB[(nt * 16 + mloc) * 17 + cd];
      }
      const v16bf bF = __builtin_bit_cast(v16bf, bu);
      acc[nt] = __builtin_amdgcn_wmma_f32_16x16x32_bf16(
          false, aF, false, bF, (short)0, acc[nt], false, false);
    }
    __syncthreads();   // all waves done with buf[ib] before it is re-DMAd
  }

  // epilogue: C layout lane[0..15]:N=lane,M=r ; lane[16..31]:N=lane-16,M=8+r
#pragma unroll
  for (int nt = 0; nt < 4; ++nt) {
#pragma unroll
    for (int r = 0; r < 8; ++r) {
      const int m = bm + wave * 16 + khal * 8 + r;
      const int n = bn + nt * 16 + mloc;
      if (m < M && n < N) {
        float cv = acc[nt][r] + (bias ? bias[n] : 0.f);
        if (ACT == 1)      cv = tanhf(cv);
        else if (ACT == 2) cv = fmaxf(cv, 0.f);
        C[(long long)m * ldc + n] = cv;
      }
    }
  }
}

// ======================================================================
// Elementwise / helper kernels
// ======================================================================
__global__ void cn_zero(float* p, long long n) {
  long long i = (long long)blockIdx.x * blockDim.x + threadIdx.x;
  if (i < n) p[i] = 0.f;
}
__global__ void cn_zero16(ushort_t* p, long long n) {
  long long i = (long long)blockIdx.x * blockDim.x + threadIdx.x;
  if (i < n) p[i] = 0;
}
__global__ void cn_copy(float* d, const float* s, long long n) {
  long long i = (long long)blockIdx.x * blockDim.x + threadIdx.x;
  if (i < n) d[i] = s[i];
}
__global__ void cn_cvt16(ushort_t* __restrict__ d, const float* __restrict__ s, long long n) {
  long long i = (long long)blockIdx.x * blockDim.x + threadIdx.x;
  if (i < n) d[i] = cn_f32_bf16(s[i]);
}
// gather rows (f32 + bf16 mirror)
__global__ void cn_gather_dual(float* __restrict__ d, ushort_t* __restrict__ d16,
                               const float* __restrict__ s, const int* __restrict__ idx,
                               int rows, int width) {
  long long i = (long long)blockIdx.x * blockDim.x + threadIdx.x;
  if (i >= (long long)rows * width) return;
  int r = (int)(i / width), c = (int)(i % width);
  float v = s[(long long)idx[r] * width + c];
  d[i] = v;  d16[i] = cn_f32_bf16(v);
}
__global__ void cn_scatter_rows(float* __restrict__ d, const float* __restrict__ s,
                                const int* __restrict__ idx, int rows, int width) {
  long long i = (long long)blockIdx.x * blockDim.x + threadIdx.x;
  if (i >= (long long)rows * width) return;
  int r = (int)(i / width), c = (int)(i % width);
  d[(long long)idx[r] * width + c] = s[i];
}
// x slice (fwd: column t; bwd: clip(len-1-t,0,L-1)) -> bf16
__global__ void cn_xslice(ushort_t* __restrict__ xt, const float* __restrict__ x,
                          const int* __restrict__ lens, int t, int rev) {
  long long i = (long long)blockIdx.x * blockDim.x + threadIdx.x;
  if (i >= (long long)C_B * C_D) return;
  int b = (int)(i / C_D), d = (int)(i % C_D);
  int j = t;
  if (rev) {
    j = lens[b] - 1 - t;
    j = j < 0 ? 0 : (j > C_L - 1 ? C_L - 1 : j);
  }
  xt[i] = cn_f32_bf16(x[((long long)b * C_L + j) * C_D + d]);
}
// GRU gate update (biases folded into xg/hg); writes h (f32) + h16 (bf16)
__global__ void cn_gru_gate(float* __restrict__ h, ushort_t* __restrict__ h16,
                            const float* __restrict__ xg, const float* __restrict__ hg,
                            const int* __restrict__ lens, int t) {
  long long i = (long long)blockIdx.x * blockDim.x + threadIdx.x;
  if (i >= (long long)C_B * C_D) return;
  int b = (int)(i / C_D), d = (int)(i % C_D);
  const float* xgb = xg + (long long)b * C_3D;
  const float* hgb = hg + (long long)b * C_3D;
  float r = cn_sigmoid(xgb[d] + hgb[d]);
  float z = cn_sigmoid(xgb[C_D + d] + hgb[C_D + d]);
  float n = tanhf(xgb[2 * C_D + d] + r * hgb[2 * C_D + d]);
  float hv = h[i];
  float hn = (1.f - z) * n + z * hv;
  float res = (t < lens[b]) ? hn : hv;
  h[i] = res;  h16[i] = cn_f32_bf16(res);
}
__global__ void cn_add2_dual(float* __restrict__ d, ushort_t* __restrict__ d16,
                             const float* __restrict__ a, const float* __restrict__ b, long long n) {
  long long i = (long long)blockIdx.x * blockDim.x + threadIdx.x;
  if (i < n) { float v = a[i] + b[i]; d[i] = v; d16[i] = cn_f32_bf16(v); }
}
// z = [facts*q, facts*m, |facts-q|, |facts-m|]  per (v,k) row -> bf16 (GEMM-only consumer)
__global__ void cn_build_z(ushort_t* __restrict__ z, const float* __restrict__ facts,
                           const float* __restrict__ q, const float* __restrict__ m) {
  long long i = (long long)blockIdx.x * blockDim.x + threadIdx.x;
  if (i >= (long long)C_B * C_D) return;
  int vk = (int)(i / C_D), d = (int)(i % C_D);
  int v = vk / C_K;
  float f  = facts[i];
  float qq = q[(long long)v * C_D + d];
  float mm = m[(long long)v * C_D + d];
  ushort_t* zr = z + (long long)vk * (4 * C_D);
  zr[d]           = cn_f32_bf16(f * qq);
  zr[C_D + d]     = cn_f32_bf16(f * mm);
  zr[2 * C_D + d] = cn_f32_bf16(fabsf(f - qq));
  zr[3 * C_D + d] = cn_f32_bf16(fabsf(f - mm));
}
// scores[row] = h1[row,:] . w2 + b2   (one wave per row)
__global__ __launch_bounds__(256)
void cn_rowdot(const float* __restrict__ H1, const float* __restrict__ w2,
               const float* __restrict__ b2, float* __restrict__ sc, int rows, int width) {
  int wave = threadIdx.x >> 5, lane = threadIdx.x & 31;
  int row = blockIdx.x * 8 + wave;
  if (row >= rows) return;
  float s = 0.f;
  for (int c = lane; c < width; c += 32) s += H1[(long long)row * width + c] * w2[c];
#pragma unroll
  for (int off = 16; off > 0; off >>= 1) s += __shfl_xor(s, off, 32);
  if (lane == 0) sc[row] = s + b2[0];
}
__global__ void cn_softmax8(const float* __restrict__ sc, float* __restrict__ g, int nv) {
  int v = blockIdx.x * blockDim.x + threadIdx.x;
  if (v >= nv) return;
  float sv[C_K], mx = -1e30f;
#pragma unroll
  for (int k = 0; k < C_K; ++k) { sv[k] = sc[v * C_K + k]; mx = fmaxf(mx, sv[k]); }
  float s = 0.f;
#pragma unroll
  for (int k = 0; k < C_K; ++k) { sv[k] = __expf(sv[k] - mx); s += sv[k]; }
  float inv = 1.f / s;
#pragma unroll
  for (int k = 0; k < C_K; ++k) g[v * C_K + k] = sv[k] * inv;
}
// attention-GRU gate; FR/FH hold facts@W^T + bias.  Writes attn f32 + bf16.
__global__ void cn_agru_gate(float* __restrict__ h, ushort_t* __restrict__ h16,
                             const float* __restrict__ FR, const float* __restrict__ FH,
                             const float* __restrict__ hur, const float* __restrict__ huh,
                             const float* __restrict__ g, int k) {
  long long i = (long long)blockIdx.x * blockDim.x + threadIdx.x;
  if (i >= (long long)C_NV * C_D) return;
  int v = (int)(i / C_D), d = (int)(i % C_D);
  long long row = ((long long)v * C_K + k) * C_D + d;
  float r  = cn_sigmoid(FR[row] + hur[i]);
  float ht = tanhf(FH[row] + r * huh[i]);
  float gg = g[v * C_K + k];
  float res = gg * ht + (1.f - gg) * h[i];
  h[i] = res;  h16[i] = cn_f32_bf16(res);
}
__global__ void cn_pack3_16(ushort_t* __restrict__ d, const float* a, const float* b,
                            const float* c, int rows) {
  long long i = (long long)blockIdx.x * blockDim.x + threadIdx.x;
  if (i >= (long long)rows * (3 * C_D)) return;
  int r = (int)(i / (3 * C_D)), col = (int)(i % (3 * C_D));
  float v;
  if (col < C_D)           v = a[(long long)r * C_D + col];
  else if (col < 2 * C_D)  v = b[(long long)r * C_D + col - C_D];
  else                     v = c[(long long)r * C_D + col - 2 * C_D];
  d[i] = cn_f32_bf16(v);
}
__global__ void cn_pack2_16(ushort_t* __restrict__ d, const float* a, int wa,
                            const float* b, int wb, int rows) {
  int w = wa + wb;
  long long i = (long long)blockIdx.x * blockDim.x + threadIdx.x;
  if (i >= (long long)rows * w) return;
  int r = (int)(i / w), col = (int)(i % w);
  float v = (col < wa) ? a[(long long)r * wa + col] : b[(long long)r * wb + (col - wa)];
  d[i] = cn_f32_bf16(v);
}

// ======================================================================
// Host-side helpers
// ======================================================================
static inline void launch_gemm(const ushort_t* A, int lda, const ushort_t* W, const float* bias,
                               float* C, int ldc, int M, int N, int Kd, int act,
                               hipStream_t s) {
  dim3 g((N + 63) / 64, (M + 63) / 64), b(128);
  switch (act) {
    case 1:  hipLaunchKernelGGL((cn_gemm_wmma<1>), g, b, LDS_TOTAL_B, s, A, lda, W, bias, C, ldc, M, N, Kd); break;
    case 2:  hipLaunchKernelGGL((cn_gemm_wmma<2>), g, b, LDS_TOTAL_B, s, A, lda, W, bias, C, ldc, M, N, Kd); break;
    default: hipLaunchKernelGGL((cn_gemm_wmma<0>), g, b, LDS_TOTAL_B, s, A, lda, W, bias, C, ldc, M, N, Kd); break;
  }
}
static inline dim3 gridFor(long long n) { return dim3((unsigned)((n + 255) / 256)); }

extern "C" void kernel_launch(void* const* d_in, const int* in_sizes, int n_in,
                              void* d_out, int out_size, void* d_ws, size_t ws_size,
                              hipStream_t stream) {
  const float* pooled = (const float*)d_in[0];
  const float* cemb   = (const float*)d_in[1];
  const int*   lens   = (const int*)  d_in[2];
  const int*   vidx   = (const int*)  d_in[3];
  const float* Wih    = (const float*)d_in[4];
  const float* Whh    = (const float*)d_in[5];
  const float* bih    = (const float*)d_in[6];
  const float* bhh    = (const float*)d_in[7];
  const float* Wq_w   = (const float*)d_in[8];
  const float* Wq_b   = (const float*)d_in[9];
  const float* W1_w   = (const float*)d_in[10];
  const float* W1_b   = (const float*)d_in[11];
  const float* W2_w   = (const float*)d_in[12];
  const float* W2_b   = (const float*)d_in[13];
  const float* aWr    = (const float*)d_in[14];
  const float* aUr    = (const float*)d_in[15];
  const float* abr    = (const float*)d_in[16];
  const float* aWh    = (const float*)d_in[17];
  const float* aUh    = (const float*)d_in[18];
  const float* abh    = (const float*)d_in[19];
  const float* nm_w   = (const float*)d_in[20];
  const float* nm_b   = (const float*)d_in[21];
  const float* up_w   = (const float*)d_in[22];
  const float* up_b   = (const float*)d_in[23];
  float* out = (float*)d_out;
  (void)ws_size; (void)in_sizes; (void)n_in; (void)out_size;

  // ---- workspace carving (bytes; keep 4B alignment) ----
  unsigned char* wsb = (unsigned char*)d_ws;
  size_t o = 0;
  auto allocf = [&](size_t n) { float* p = (float*)(wsb + o); o += n * 4; return p; };
  auto alloch = [&](size_t n) { ushort_t* p = (ushort_t*)(wsb + o); o += ((n * 2 + 3) & ~(size_t)3); return p; };

  // f32 buffers
  float* big0  = allocf((size_t)C_B * C_3D);   // GRU hg  | mem-phase: z16 view
  float* big1  = allocf((size_t)C_B * C_3D);   // GRU xg  | mem-phase: h1
  float* hf    = allocf((size_t)C_B * C_D);
  float* hb    = allocf((size_t)C_B * C_D);
  float* facts = allocf((size_t)C_B * C_D);
  float* feats = allocf((size_t)C_NV * C_F);
  float* qbuf  = allocf((size_t)C_NV * C_D);
  float* prem  = allocf((size_t)C_NV * C_D);
  float* FR    = allocf((size_t)C_B * C_D);
  float* FH    = allocf((size_t)C_B * C_D);
  float* hur   = allocf((size_t)C_NV * C_D);
  float* huh   = allocf((size_t)C_NV * C_D);
  float* attn  = allocf((size_t)C_NV * C_D);
  float* sc    = allocf((size_t)C_B);
  float* gco   = allocf((size_t)C_B);
  float* upd   = allocf((size_t)C_NV * C_F);
  // bf16 weight mirrors
  ushort_t* wih16 = alloch((size_t)2 * C_3D * C_D);
  ushort_t* whh16 = alloch((size_t)2 * C_3D * C_D);
  ushort_t* wq16  = alloch((size_t)C_D * C_F);
  ushort_t* w116  = alloch((size_t)C_F * 4 * C_D);
  ushort_t* wr16  = alloch((size_t)C_D * C_D);
  ushort_t* ur16  = alloch((size_t)C_D * C_D);
  ushort_t* wh16w = alloch((size_t)C_D * C_D);
  ushort_t* uh16  = alloch((size_t)C_D * C_D);
  ushort_t* nm16  = alloch((size_t)C_D * C_3D);
  ushort_t* up16  = alloch((size_t)C_F * (C_F + C_D));
  // bf16 activation mirrors
  ushort_t* xt16    = alloch((size_t)C_B * C_D);
  ushort_t* h16     = alloch((size_t)C_B * C_D);
  ushort_t* facts16 = alloch((size_t)C_B * C_D);
  ushort_t* feats16 = alloch((size_t)C_NV * C_F);
  ushort_t* attn16  = alloch((size_t)C_NV * C_D);
  ushort_t* cat916  = alloch((size_t)C_NV * C_3D);
  ushort_t* cat816  = alloch((size_t)C_NV * (C_F + C_D));

  ushort_t* z16   = (ushort_t*)big0;                 // C_B x 1200 bf16 (39.3MB < big0)
  float*    h1buf = big1;                            // C_B x 512 f32 (33.5MB < big1)
  float*    hgbuf = big0;                            // GRU-phase aliases
  float*    xgbuf = big1;

  // ---------- 0) one-time weight conversion to bf16 ----------
  cn_cvt16<<<gridFor(2LL * C_3D * C_D), 256, 0, stream>>>(wih16, Wih, 2LL * C_3D * C_D);
  cn_cvt16<<<gridFor(2LL * C_3D * C_D), 256, 0, stream>>>(whh16, Whh, 2LL * C_3D * C_D);
  cn_cvt16<<<gridFor((long long)C_D * C_F), 256, 0, stream>>>(wq16, Wq_w, (long long)C_D * C_F);
  cn_cvt16<<<gridFor((long long)C_F * 4 * C_D), 256, 0, stream>>>(w116, W1_w, (long long)C_F * 4 * C_D);
  cn_cvt16<<<gridFor((long long)C_D * C_D), 256, 0, stream>>>(wr16, aWr, (long long)C_D * C_D);
  cn_cvt16<<<gridFor((long long)C_D * C_D), 256, 0, stream>>>(ur16, aUr, (long long)C_D * C_D);
  cn_cvt16<<<gridFor((long long)C_D * C_D), 256, 0, stream>>>(wh16w, aWh, (long long)C_D * C_D);
  cn_cvt16<<<gridFor((long long)C_D * C_D), 256, 0, stream>>>(uh16, aUh, (long long)C_D * C_D);
  cn_cvt16<<<gridFor((long long)C_D * C_3D), 256, 0, stream>>>(nm16, nm_w, (long long)C_D * C_3D);
  cn_cvt16<<<gridFor((long long)C_F * (C_F + C_D)), 256, 0, stream>>>(up16, up_w, (long long)C_F * (C_F + C_D));

  // ---------- 1) feats_v = pooled[valid_obj_idx] ; q = tanh(feats @ Wq^T + b) ----------
  cn_gather_dual<<<gridFor((long long)C_NV * C_F), 256, 0, stream>>>(feats, feats16, pooled, vidx, C_NV, C_F);
  launch_gemm(feats16, C_F, wq16, Wq_b, qbuf, C_D, C_NV, C_D, C_F, /*tanh*/1, stream);

  // ---------- 2) bidirectional GRU over 16384 sequences ----------
  for (int dir = 0; dir < 2; ++dir) {
    float* h = dir ? hb : hf;
    cn_zero<<<gridFor((long long)C_B * C_D), 256, 0, stream>>>(h, (long long)C_B * C_D);
    cn_zero16<<<gridFor((long long)C_B * C_D), 256, 0, stream>>>(h16, (long long)C_B * C_D);
    const ushort_t* Wi = wih16 + (size_t)dir * C_3D * C_D;
    const ushort_t* Wh = whh16 + (size_t)dir * C_3D * C_D;
    const float*    bi = bih + (size_t)dir * C_3D;
    const float*    bh = bhh + (size_t)dir * C_3D;
    for (int t = 0; t < C_L; ++t) {
      cn_xslice<<<gridFor((long long)C_B * C_D), 256, 0, stream>>>(xt16, cemb, lens, t, dir);
      launch_gemm(xt16, C_D, Wi, bi, xgbuf, C_3D, C_B, C_3D, C_D, 0, stream); // x_t @ Wih^T + bih
      launch_gemm(h16,  C_D, Wh, bh, hgbuf, C_3D, C_B, C_3D, C_D, 0, stream); // h   @ Whh^T + bhh
      cn_gru_gate<<<gridFor((long long)C_B * C_D), 256, 0, stream>>>(h, h16, xgbuf, hgbuf, lens, t);
    }
  }
  cn_add2_dual<<<gridFor((long long)C_B * C_D), 256, 0, stream>>>(facts, facts16, hf, hb, (long long)C_B * C_D);

  // ---------- 3) FR = facts@Wr^T+br, FH = facts@Wh^T+bh (reused across T_M) ----------
  launch_gemm(facts16, C_D, wr16, abr, FR, C_D, C_B, C_D, C_D, 0, stream);
  launch_gemm(facts16, C_D, wh16w, abh, FH, C_D, C_B, C_D, C_D, 0, stream);

  // ---------- 4) episodic memory loop ----------
  cn_copy<<<gridFor((long long)C_NV * C_D), 256, 0, stream>>>(prem, qbuf, (long long)C_NV * C_D);
  for (int it = 0; it < C_TM; ++it) {
    cn_build_z<<<gridFor((long long)C_B * C_D), 256, 0, stream>>>(z16, facts, qbuf, prem);
    launch_gemm(z16, 4 * C_D, w116, W1_b, h1buf, C_F, C_B, C_F, 4 * C_D, /*tanh*/1, stream);
    cn_rowdot<<<(C_B + 7) / 8, 256, 0, stream>>>(h1buf, W2_w, W2_b, sc, C_B, C_F);
    cn_softmax8<<<gridFor(C_NV), 256, 0, stream>>>(sc, gco, C_NV);

    cn_zero<<<gridFor((long long)C_NV * C_D), 256, 0, stream>>>(attn, (long long)C_NV * C_D);
    cn_zero16<<<gridFor((long long)C_NV * C_D), 256, 0, stream>>>(attn16, (long long)C_NV * C_D);
    for (int k = 0; k < C_K; ++k) {
      launch_gemm(attn16, C_D, ur16, nullptr, hur, C_D, C_NV, C_D, C_D, 0, stream); // h @ Ur^T
      launch_gemm(attn16, C_D, uh16, nullptr, huh, C_D, C_NV, C_D, C_D, 0, stream); // h @ Uh^T
      cn_agru_gate<<<gridFor((long long)C_NV * C_D), 256, 0, stream>>>(attn, attn16, FR, FH, hur, huh, gco, k);
    }
    cn_pack3_16<<<gridFor((long long)C_NV * C_3D), 256, 0, stream>>>(cat916, prem, attn, qbuf, C_NV);
    launch_gemm(cat916, C_3D, nm16, nm_b, prem, C_D, C_NV, C_D, C_3D, /*relu*/2, stream);
  }

  // ---------- 5) object update + scatter ----------
  cn_pack2_16<<<gridFor((long long)C_NV * (C_F + C_D)), 256, 0, stream>>>(cat816, feats, C_F, prem, C_D, C_NV);
  launch_gemm(cat816, C_F + C_D, up16, up_b, upd, C_F, C_NV, C_F, C_F + C_D, /*relu*/2, stream);

  cn_copy<<<gridFor((long long)C_NOBJ * C_F), 256, 0, stream>>>(out, pooled, (long long)C_NOBJ * C_F);
  cn_scatter_rows<<<gridFor((long long)C_NV * C_F), 256, 0, stream>>>(out, upd, vidx, C_NV, C_F);
}
